// GroupedQueryAttention_23691039605412
// MI455X (gfx1250) — matrix operand order repaired and checked by reading, not compiled, not run on hardware
//
#include <hip/hip_runtime.h>
#include <hip/hip_bf16.h>

// ---------------------------------------------------------------------------
// GroupedQueryAttention for MI455X (gfx1250, wave32, WMMA 16x16x32 f16).
//   B=2, T=2048, D=2048, H=32, KVH=8, HEAD_DIM=64, REP=4.
// Pipeline: f32->f16 convert -> Q/K/V GEMM (WMMA) -> flash attention (WMMA)
//           -> O GEMM (WMMA, f32 out).
// Tile staging uses GLOBAL_LOAD_ASYNC_TO_LDS_B128 (ASYNCcnt) when available.
// ---------------------------------------------------------------------------

typedef __attribute__((ext_vector_type(16))) _Float16 v16h;
typedef __attribute__((ext_vector_type(8)))  float    v8f;
typedef int v4i __attribute__((vector_size(16)));

union F16x16 { v16h v; uint4 u[2]; };

__device__ __forceinline__ v8f wmma16x16x32(v16h a, v16h b, v8f c) {
  // D = A(16x32) * B(32x16) + C, fp32 accumulate.
  return __builtin_amdgcn_wmma_f32_16x16x32_f16(
      /*neg_a=*/false, a, /*neg_b=*/false, b,
      /*c_mod=*/(short)0, c, /*reuse_a=*/false, /*reuse_b=*/false);
}

// ---- async memory->LDS staging (CDNA5), with synchronous fallback ---------
#define AS1 __attribute__((address_space(1)))
#define AS3 __attribute__((address_space(3)))
#if __has_builtin(__builtin_amdgcn_global_load_async_to_lds_b128) && \
    __has_builtin(__builtin_amdgcn_s_wait_asynccnt)
#define HAVE_ASYNC_LDS 1
#else
#define HAVE_ASYNC_LDS 0
#endif

__device__ __forceinline__ void copy16_g2lds(const _Float16* g, _Float16* l) {
#if HAVE_ASYNC_LDS
  __builtin_amdgcn_global_load_async_to_lds_b128(
      (AS1 v4i*)(void*)g, (AS3 v4i*)(void*)l, /*offset=*/0, /*cpol=*/0);
#else
  *(uint4*)l = *(const uint4*)g;
#endif
}
__device__ __forceinline__ void async_lds_fence() {
#if HAVE_ASYNC_LDS
  __builtin_amdgcn_s_wait_asynccnt(0);
#endif
}

// Model constants
#define TSEQ   2048
#define DMODEL 2048
#define NHEAD  32
#define NKV    8
#define HDIM   64
#define BATCH  2

// ---------------------------------------------------------------------------
// fp32 -> fp16 conversion (grid-stride)
// ---------------------------------------------------------------------------
__global__ void cvt_f32_to_f16(const float* __restrict__ src,
                               _Float16* __restrict__ dst, int n) {
  for (int i = blockIdx.x * blockDim.x + threadIdx.x; i < n;
       i += gridDim.x * blockDim.x)
    dst[i] = (_Float16)src[i];
}

// ---------------------------------------------------------------------------
// GEMM: C[M,N] = (A[M,K] @ W[K,N] + bias[N]) * scale
//   A,W: f16 row-major.  C: f16 or f32 (template).
//   Block: 128 thr (4 waves). Block tile 128x64, wave tile 32x64, K step 32.
//   8 WMMAs per wave per K-chunk from 2 A-frags x 4 B-frags.
// ---------------------------------------------------------------------------
#define GTM 128
#define GTN 64
#define GTK 32

template <bool OUT_F16>
__global__ __launch_bounds__(128)
void gemm_wmma_f16(const _Float16* __restrict__ A, const _Float16* __restrict__ W,
                   const float* __restrict__ bias, void* __restrict__ Cout,
                   int M, int N, int K, float scale) {
  __shared__ __align__(16) _Float16 As[GTM][GTK];   // 8 KB
  __shared__ __align__(16) _Float16 Wt[GTN][GTK];   // 4 KB, W transposed (n-major)
  const int tid  = threadIdx.x;
  const int lane = tid & 31;
  const int wave = tid >> 5;
  const int n0 = blockIdx.x * GTN;
  const int m0 = blockIdx.y * GTM;

  v8f acc[2][4] = {};

  for (int k0 = 0; k0 < K; k0 += GTK) {
    // Stage A tile: 128 rows x 32 halfs via async 16B lane-copies (no VGPR).
    for (int c = tid; c < (GTM * GTK) / 8; c += 128) {
      int r = c >> 2, cc = c & 3;
      copy16_g2lds(&A[(size_t)(m0 + r) * K + k0 + cc * 8], &As[r][cc * 8]);
    }
    // Stage W tile transposed: Wt[n][k] = W[k0+k][n0+n] (coalesced reads).
    for (int e = tid; e < GTK * GTN; e += 128) {
      int kk = e >> 6, nn = e & 63;
      Wt[nn][kk] = W[(size_t)(k0 + kk) * N + n0 + nn];
    }
    if (k0 + GTK < K)  // warm next A tile (global_prefetch_b8)
      __builtin_prefetch(&A[(size_t)(m0 + tid) * K + k0 + GTK], 0, 1);
    async_lds_fence();
    __syncthreads();

    // A fragments (16x32, doc layout: lane<16 K={0..7,16..23}, lane>=16 +8).
    F16x16 af[2];
    const int kb = (lane < 16) ? 0 : 8;
#pragma unroll
    for (int mt = 0; mt < 2; ++mt) {
      const _Float16* arow = &As[wave * 32 + mt * 16 + (lane & 15)][0];
      af[mt].u[0] = *(const uint4*)(arow + kb);
      af[mt].u[1] = *(const uint4*)(arow + kb + 16);
    }
    // 4 B fragments (32x16: N=lane%16, lane<16 K={0..7,16..23}, lane>=16 +8),
    // each reused across both A fragments.
    const int kbb = (lane < 16) ? 0 : 8;
#pragma unroll
    for (int nt = 0; nt < 4; ++nt) {
      F16x16 bf;
      const _Float16* bcol = &Wt[nt * 16 + (lane & 15)][kbb];
      bf.u[0] = *(const uint4*)(bcol);
      bf.u[1] = *(const uint4*)(bcol + 16);
      acc[0][nt] = wmma16x16x32(af[0].v, bf.v, acc[0][nt]);
      acc[1][nt] = wmma16x16x32(af[1].v, bf.v, acc[1][nt]);
    }
    __syncthreads();
  }

  // Epilogue. C layout: row = r + (lane>=16 ? 8 : 0), col = lane%16.
  const int mhi = (lane < 16) ? 0 : 8;
#pragma unroll
  for (int mt = 0; mt < 2; ++mt) {
    const int mrow = m0 + wave * 32 + mt * 16 + mhi;
#pragma unroll
    for (int nt = 0; nt < 4; ++nt) {
      const int ncol = n0 + nt * 16 + (lane & 15);
      const float bv = bias[ncol];
#pragma unroll
      for (int r = 0; r < 8; ++r) {
        float v = (acc[mt][nt][r] + bv) * scale;
        size_t idx = (size_t)(mrow + r) * N + ncol;
        if (OUT_F16) ((_Float16*)Cout)[idx] = (_Float16)v;
        else         ((float*)Cout)[idx] = v;
      }
    }
  }
}

// ---------------------------------------------------------------------------
// Flash attention core (causal, GQA).
//   Q: [B,T,H,64] f16 (pre-scaled by 1/sqrt(64)), K/V: [B,T,KVH,64] f16.
//   O: [B,T,H,64] f16.
//   Block: 128 thr, one (b,h) and 64 query rows; wave w -> 16-row q tile.
//   S loop: 32 keys per tile; QK^T and PV via WMMA; online softmax in
//   the WMMA C-register layout (per-row stats via xor-shuffles per half).
// ---------------------------------------------------------------------------
__global__ __launch_bounds__(128)
void gqa_flash_attn(const _Float16* __restrict__ Q, const _Float16* __restrict__ Kh,
                    const _Float16* __restrict__ Vh, _Float16* __restrict__ O) {
  __shared__ __align__(16) _Float16 Ks[32][64];      // keys x dims   (4 KB)
  __shared__ __align__(16) _Float16 Vt[64][32];      // dims x keys   (4 KB)
  __shared__ __align__(16) _Float16 Pt[4][16][32];   // per-wave P    (4 KB)

  const int tid  = threadIdx.x;
  const int lane = tid & 31;
  const int wave = tid >> 5;
  const int bh = blockIdx.y;                 // 0..B*NHEAD-1
  const int b  = bh >> 5;
  const int h  = bh & 31;
  const int g  = h >> 2;                     // kv group (REP=4)
  const int q0blk = blockIdx.x * 64;
  const int q0 = q0blk + wave * 16;

  // Q A-fragments: 16 rows x 64 dims -> two 16x32 frags, straight from global.
  F16x16 qa[2];
  {
    const _Float16* qrow =
        Q + ((size_t)(b * TSEQ + q0 + (lane & 15)) * NHEAD + h) * HDIM;
    const int kb = (lane < 16) ? 0 : 8;
#pragma unroll
    for (int c = 0; c < 2; ++c) {
      qa[c].u[0] = *(const uint4*)(qrow + c * 32 + kb);
      qa[c].u[1] = *(const uint4*)(qrow + c * 32 + kb + 16);
    }
  }

  v8f acc[4] = {};                            // 16x64 output accumulator
  float mrow[8], lrow[8];
#pragma unroll
  for (int r = 0; r < 8; ++r) { mrow[r] = -1e30f; lrow[r] = 0.f; }

  const int Mhi = (lane < 16) ? 0 : 8;        // C-layout row offset
  const int nstiles = q0blk / 32 + 2;         // causal: keys s <= q0blk+63

  for (int st = 0; st < nstiles; ++st) {
    const int s0 = st * 32;
    // ---- stage K tile (row-major) via async 16B lane-copies ----
    for (int c = tid; c < (32 * 64) / 8; c += 128) {
      int r = c >> 3, cc = c & 7;
      copy16_g2lds(Kh + ((size_t)(b * TSEQ + s0 + r) * NKV + g) * HDIM + cc * 8,
                   &Ks[r][cc * 8]);
    }
    // ---- stage V tile transposed: Vt[d][s] ----
    for (int e = tid; e < 32 * 64; e += 128) {
      int s = e >> 6, d = e & 63;
      Vt[d][s] = Vh[((size_t)(b * TSEQ + s0 + s) * NKV + g) * HDIM + d];
    }
    async_lds_fence();
    __syncthreads();

    if (s0 <= q0 + 15) {                      // wave-uniform skip past diagonal
      // ---- scores: S(16x32) = Q(16x64) @ K^T(64x32), two 16x16 tiles ----
      v8f sc[2] = {};
#pragma unroll
      for (int t = 0; t < 2; ++t) {
#pragma unroll
        for (int c = 0; c < 2; ++c) {
          F16x16 bf;  // B = K^T: N=key=lane%16, K=dim contiguous in Ks row
          const _Float16* kr =
              &Ks[t * 16 + (lane & 15)][c * 32 + ((lane < 16) ? 0 : 8)];
          bf.u[0] = *(const uint4*)kr;
          bf.u[1] = *(const uint4*)(kr + 16);
          sc[t] = wmma16x16x32(qa[c].v, bf.v, sc[t]);
        }
      }
      // ---- causal mask + per-row max (xor-shuffle over each 16-lane half) --
      float tmax[8];
#pragma unroll
      for (int r = 0; r < 8; ++r) tmax[r] = -1e30f;
#pragma unroll
      for (int t = 0; t < 2; ++t) {
        const int s = s0 + t * 16 + (lane & 15);
#pragma unroll
        for (int r = 0; r < 8; ++r) {
          float v = sc[t][r];
          if (s > q0 + r + Mhi) v = -1e30f;
          sc[t][r] = v;
          tmax[r] = fmaxf(tmax[r], v);
        }
      }
#pragma unroll
      for (int off = 8; off >= 1; off >>= 1)
#pragma unroll
        for (int r = 0; r < 8; ++r)
          tmax[r] = fmaxf(tmax[r], __shfl_xor(tmax[r], off, 32));

      // ---- online rescale of accumulator + stats ----
      float psum[8];
#pragma unroll
      for (int r = 0; r < 8; ++r) {
        const float mnew = fmaxf(mrow[r], tmax[r]);
        const float scl = __expf(mrow[r] - mnew);
        mrow[r] = mnew;
        lrow[r] *= scl;
#pragma unroll
        for (int dt = 0; dt < 4; ++dt) acc[dt][r] *= scl;
        psum[r] = 0.f;
      }
      // ---- P = exp(S - m); spill f16 P to per-wave LDS (C -> A relayout) --
#pragma unroll
      for (int t = 0; t < 2; ++t) {
#pragma unroll
        for (int r = 0; r < 8; ++r) {
          const float p = __expf(sc[t][r] - mrow[r]);
          psum[r] += p;
          Pt[wave][r + Mhi][t * 16 + (lane & 15)] = (_Float16)p;
        }
      }
#pragma unroll
      for (int off = 8; off >= 1; off >>= 1)
#pragma unroll
        for (int r = 0; r < 8; ++r)
          psum[r] += __shfl_xor(psum[r], off, 32);
#pragma unroll
      for (int r = 0; r < 8; ++r) lrow[r] += psum[r];

      // ---- O += P(16x32) @ V(32x64): 4 WMMAs ----
      F16x16 pa;  // in-wave LDS ops are kept in order by hardware
      {
        const _Float16* prow = &Pt[wave][lane & 15][0];
        const int kb = (lane < 16) ? 0 : 8;
        pa.u[0] = *(const uint4*)(prow + kb);
        pa.u[1] = *(const uint4*)(prow + kb + 16);
      }
#pragma unroll
      for (int dt = 0; dt < 4; ++dt) {
        F16x16 bf;  // B = V: N=dim=lane%16, K=key contiguous in Vt row
        const _Float16* vr = &Vt[dt * 16 + (lane & 15)][(lane < 16) ? 0 : 8];
        bf.u[0] = *(const uint4*)vr;
        bf.u[1] = *(const uint4*)(vr + 16);
        acc[dt] = wmma16x16x32(pa.v, bf.v, acc[dt]);
      }
    }
    __syncthreads();
  }

  // ---- normalize + write O (f16) ----
#pragma unroll
  for (int dt = 0; dt < 4; ++dt) {
#pragma unroll
    for (int r = 0; r < 8; ++r) {
      const float v = acc[dt][r] / lrow[r];
      const int qr = q0 + r + Mhi;
      O[((size_t)(b * TSEQ + qr) * NHEAD + h) * HDIM + dt * 16 + (lane & 15)] =
          (_Float16)v;
    }
  }
}

// ---------------------------------------------------------------------------
// Host-side launcher
// ---------------------------------------------------------------------------
extern "C" void kernel_launch(void* const* d_in, const int* in_sizes, int n_in,
                              void* d_out, int out_size, void* d_ws, size_t ws_size,
                              hipStream_t stream) {
  (void)in_sizes; (void)n_in; (void)out_size; (void)ws_size;
  const float* x  = (const float*)d_in[0];
  // d_in[1] = mask (causal; computed analytically in-kernel)
  const float* wq = (const float*)d_in[2];
  const float* bq = (const float*)d_in[3];
  const float* wk = (const float*)d_in[4];
  const float* bk = (const float*)d_in[5];
  const float* wv = (const float*)d_in[6];
  const float* bv = (const float*)d_in[7];
  const float* wo = (const float*)d_in[8];
  const float* bo = (const float*)d_in[9];
  float* out = (float*)d_out;

  const int M = BATCH * TSEQ;                       // 4096
  // Workspace carve-up (f16), ~80 MB total -> fits 192 MB L2.
  _Float16* ws   = (_Float16*)d_ws;
  _Float16* x_h  = ws;
  _Float16* wq_h = x_h  + (size_t)M * DMODEL;
  _Float16* wk_h = wq_h + (size_t)DMODEL * DMODEL;
  _Float16* wv_h = wk_h + (size_t)DMODEL * (NKV * HDIM);
  _Float16* wo_h = wv_h + (size_t)DMODEL * (NKV * HDIM);
  _Float16* q_h  = wo_h + (size_t)DMODEL * DMODEL;
  _Float16* k_h  = q_h  + (size_t)M * DMODEL;
  _Float16* v_h  = k_h  + (size_t)M * (NKV * HDIM);
  _Float16* ao_h = v_h  + (size_t)M * (NKV * HDIM);

  // 1) convert to f16
  cvt_f32_to_f16<<<2048, 256, 0, stream>>>(x,  x_h,  M * DMODEL);
  cvt_f32_to_f16<<<2048, 256, 0, stream>>>(wq, wq_h, DMODEL * DMODEL);
  cvt_f32_to_f16<<<1024, 256, 0, stream>>>(wk, wk_h, DMODEL * NKV * HDIM);
  cvt_f32_to_f16<<<1024, 256, 0, stream>>>(wv, wv_h, DMODEL * NKV * HDIM);
  cvt_f32_to_f16<<<2048, 256, 0, stream>>>(wo, wo_h, DMODEL * DMODEL);

  // 2) projections (scale 1/sqrt(HDIM) folded into Q)
  gemm_wmma_f16<true><<<dim3(DMODEL / GTN, M / GTM), 128, 0, stream>>>(
      x_h, wq_h, bq, q_h, M, DMODEL, DMODEL, 0.125f);
  gemm_wmma_f16<true><<<dim3((NKV * HDIM) / GTN, M / GTM), 128, 0, stream>>>(
      x_h, wk_h, bk, k_h, M, NKV * HDIM, DMODEL, 1.0f);
  gemm_wmma_f16<true><<<dim3((NKV * HDIM) / GTN, M / GTM), 128, 0, stream>>>(
      x_h, wv_h, bv, v_h, M, NKV * HDIM, DMODEL, 1.0f);

  // 3) attention: grid (T/64 q-tiles, B*H)
  gqa_flash_attn<<<dim3(TSEQ / 64, BATCH * NHEAD), 128, 0, stream>>>(
      q_h, k_h, v_h, ao_h);

  // 4) output projection, f32 result
  gemm_wmma_f16<false><<<dim3(DMODEL / GTN, M / GTM), 128, 0, stream>>>(
      ao_h, wo_h, bo, out, M, DMODEL, DMODEL, 1.0f);
}
